// PixelAttention_7249904795802
// MI455X (gfx1250) — matrix operand order repaired and checked by reading, not compiled, not run on hardware
//
#include <hip/hip_runtime.h>
#include <math.h>

// Problem constants (reference: N=8, C=256, H=8, D=32, T=32*32=1024)
#define NB   8
#define NCH  256
#define HEADS 8
#define HDIM 32
#define SEQT 1024

typedef __attribute__((ext_vector_type(16))) __bf16 v16bf;
typedef __attribute__((ext_vector_type(8)))  float  v8f;
typedef __attribute__((ext_vector_type(4)))  unsigned int v4u;

union BFrag { v16bf v; unsigned short u[16]; v4u q[2]; };
union F8    { v8f   v; float          f[8];  };

// Branchless f32 -> bf16 round-to-nearest-even (inputs are finite here).
__device__ __forceinline__ unsigned short f2bf(float x) {
  unsigned int u = __float_as_uint(x);
  u += 0x7fffu + ((u >> 16) & 1u);
  return (unsigned short)(u >> 16);
}

__device__ __forceinline__ v8f wmma_bf16(v16bf a, v16bf b, v8f c) {
  // D = A(16x32 bf16) * B(32x16 bf16) + C(16x16 f32)
  return __builtin_amdgcn_wmma_f32_16x16x32_bf16(false, a, false, b, (short)0, c, false, false);
}

// ---- Fragment-flat index maps (inverse of §7.12.2 wave32 layouts) ----
// A 16(M)x32(K) bf16: lane L: m=L%16, half=L/16; dword v: k=(v>>2)*16+half*8+(v&3)*2
__device__ __forceinline__ int flatA(int m, int k) {
  return ((m + (((k >> 3) & 1) << 4)) << 4) + ((k >> 4) << 3) + (k & 7);
}
// B 32(K)x16(N) bf16: lane L: n=L%16, half=L/16; dword v: k=(v>>2)*16+half*8+(v&3)*2
__device__ __forceinline__ int flatB(int k, int n) {
  return ((n + (((k >> 3) & 1) << 4)) << 4) + ((k >> 4) << 3) + (k & 7);
}

// Load one fragment stored in flat order: lane reads 32 contiguous bytes (2 x b128).
__device__ __forceinline__ v16bf load_frag(const unsigned short* base) {
  const v4u* p = (const v4u*)(base + ((threadIdx.x & 31) << 4));
  BFrag f;
  f.q[0] = p[0];
  f.q[1] = p[1];
  return f.v;
}

// ------------------- Kernel S: pack image (seq) to bf16 A-fragment order -------------------
// grid (64 tblk, 8 cblk, 8 n), 1 wave builds one 16x32 A fragment. Done ONCE; the
// KQV GEMM then reuses each fragment 48 times as cheap b128 loads from L2.
__global__ void __launch_bounds__(32)
pack_seq(const float* __restrict__ image, unsigned short* __restrict__ seqP) {
  const int tb = blockIdx.x, cb = blockIdx.y, n = blockIdx.z;
  // element (m,k) = image[n][cb*32+k][tb*16+m] : lanes coalesce along t (m)
  const float* src = image + ((size_t)n * NCH + (cb << 5)) * SEQT + (tb << 4);
  const int lane = threadIdx.x & 31, m = lane & 15, half = lane >> 4;
  BFrag f;
#pragma unroll
  for (int v = 0; v < 8; ++v) {
    const int k = ((v >> 2) << 4) + (half << 3) + ((v & 3) << 1);
    f.u[2 * v]     = f2bf(src[(size_t)k * SEQT + m]);
    f.u[2 * v + 1] = f2bf(src[(size_t)(k + 1) * SEQT + m]);
  }
  v4u* dst = (v4u*)(seqP + ((((size_t)((n << 6) + tb)) << 3) + cb) * 512 + (lane << 4));
  dst[0] = f.q[0];
  dst[1] = f.q[1];
}

// ------------------- Kernel W: pack weights to bf16 B-fragment order -------------------
__global__ void __launch_bounds__(256)
pack_weights(const float* __restrict__ w_kqv, const float* __restrict__ w_mix,
             unsigned short* __restrict__ wkqvP, unsigned short* __restrict__ wmixP) {
  int i = blockIdx.x * 256 + threadIdx.x;
  if (i < NCH * 3 * NCH) {                       // w_kqv[c][j], c<256, j<768
    int c = i / (3 * NCH), j = i - c * (3 * NCH);
    int tile = (c >> 5) * 48 + (j >> 4);
    wkqvP[(tile << 9) + flatB(c & 31, j & 15)] = f2bf(w_kqv[i]);
  }
  if (i < NCH * NCH) {                           // w_mix[c][j], c<256, j<256
    int c = i >> 8, j = i & 255;
    int tile = (c >> 5) * 16 + (j >> 4);
    wmixP[(tile << 9) + flatB(c & 31, j & 15)] = f2bf(w_mix[i]);
  }
}

// ------------------- Kernel 1: KQV projection + fragment-order scatter -------------------
// grid (64 tblk, 48 jblk, 8 n), 1 wave. Output tile 16(t) x 16(j). Pure frag loads + WMMA.
__global__ void __launch_bounds__(32)
kqv_gemm(const unsigned short* __restrict__ seqP, const unsigned short* __restrict__ wkqvP,
         const float* __restrict__ b_kqv,
         unsigned short* __restrict__ qP, unsigned short* __restrict__ kP,
         unsigned short* __restrict__ vP) {
  const int tb = blockIdx.x, jb = blockIdx.y, n = blockIdx.z;
  const int t0 = tb << 4, j0 = jb << 4;
  F8 acc; acc.v = (v8f){0.f, 0.f, 0.f, 0.f, 0.f, 0.f, 0.f, 0.f};
#pragma unroll
  for (int kc = 0; kc < NCH; kc += 32) {
    const int cb = kc >> 5;
    const int tileB = cb * 48 + jb;
    if (kc + 32 < NCH)
      __builtin_prefetch(wkqvP + ((size_t)(tileB + 48) << 9), 0, 0);
    v16bf a = load_frag(seqP + ((size_t)(((n << 6) + tb) << 3) + cb) * 512);
    v16bf b = load_frag(wkqvP + ((size_t)tileB << 9));
    acc.v = wmma_bf16(a, b, acc.v);
  }
  const int lane = threadIdx.x & 31, ncol = lane & 15, half = lane >> 4;
  const int j = j0 + ncol;
  const float bias = b_kqv[j];
  const int which = j >> 8;            // 0=k, 1=q, 2=v (reference split order)
  const int h = (j & 255) >> 5, d = j & 31;
#pragma unroll
  for (int r = 0; r < 8; ++r) {
    const int t = t0 + (half << 3) + r;
    const unsigned short val = f2bf(acc.f[r] + bias);
    if (which == 1) {        // Q: A-fragment tiles of 16 query rows, k-dim = d
      qP[(((size_t)((n * HEADS + h) << 6) + (t >> 4)) << 9) + flatA(t & 15, d)] = val;
    } else if (which == 0) { // K: B-fragment per 16-key tile (k-dim = d, n = key%16)
      kP[(((size_t)((n * HEADS + h) << 6) + (t >> 4)) << 9) + flatB(d, t & 15)] = val;
    } else {                 // V: B-fragment per (32-key block, d-half)
      const size_t tile = (size_t)((((n * HEADS + h) << 5) + (t >> 5)) << 1) + (d >> 4);
      vP[(tile << 9) + flatB(t & 31, d & 15)] = val;
    }
  }
}

// ------------------- Kernel 2: causal flash attention -------------------
// grid (64 qblk, 8 h, 8 n), 1 wave owns 16 query rows, streams keys in 32-chunks.
__global__ void __launch_bounds__(32)
attn_flash(const unsigned short* __restrict__ qP, const unsigned short* __restrict__ kP,
           const unsigned short* __restrict__ vP, unsigned short* __restrict__ attP) {
  __shared__ __align__(16) unsigned short pbuf[512];   // P tile in A-fragment flat order
  const int qb = blockIdx.x, h = blockIdx.y, n = blockIdx.z;
  const int q0 = qb << 4;
  const int nh = n * HEADS + h;
  const int lane = threadIdx.x & 31, ncol = lane & 15, half = lane >> 4;
  const float scale = 0.17677669529663687f;  // 1/sqrt(32)
  const float NEG = -__builtin_inff();

  v16bf qa = load_frag(qP + ((size_t)((nh << 6) + qb) << 9));  // Q 16x32 (D == WMMA K)
  F8 acc0, acc1;
  acc0.v = (v8f){0.f, 0.f, 0.f, 0.f, 0.f, 0.f, 0.f, 0.f};
  acc1.v = acc0.v;
  float mrow[8], lrow[8];
#pragma unroll
  for (int r = 0; r < 8; ++r) { mrow[r] = NEG; lrow[r] = 0.f; }

  for (int s0 = 0; s0 <= q0; s0 += 32) {      // key pair-blocks up through the diagonal
    const v8f z = (v8f){0.f, 0.f, 0.f, 0.f, 0.f, 0.f, 0.f, 0.f};
    F8 s_a, s_b;
    s_a.v = wmma_bf16(qa, load_frag(kP + ((size_t)((nh << 6) + (s0 >> 4)) << 9)), z);
    s_b.v = wmma_bf16(qa, load_frag(kP + ((size_t)((nh << 6) + (s0 >> 4) + 1) << 9)), z);

#pragma unroll
    for (int r = 0; r < 8; ++r) {
      const int row = (half << 3) + r;
      const int t = q0 + row;
      float v0 = s_a.f[r] * scale; if (s0 + ncol > t)      v0 = NEG;  // causal mask
      float v1 = s_b.f[r] * scale; if (s0 + 16 + ncol > t) v1 = NEG;
      float lm = fmaxf(v0, v1);
      lm = fmaxf(lm, __shfl_xor(lm, 1, 32));
      lm = fmaxf(lm, __shfl_xor(lm, 2, 32));
      lm = fmaxf(lm, __shfl_xor(lm, 4, 32));
      lm = fmaxf(lm, __shfl_xor(lm, 8, 32));  // row max within 16-lane half
      const float mnew = fmaxf(mrow[r], lm);
      const float alpha = __expf(mrow[r] - mnew);
      const float p0 = __expf(v0 - mnew);
      const float p1 = __expf(v1 - mnew);
      float rs = p0 + p1;
      rs += __shfl_xor(rs, 1, 32);
      rs += __shfl_xor(rs, 2, 32);
      rs += __shfl_xor(rs, 4, 32);
      rs += __shfl_xor(rs, 8, 32);            // row sum within half
      lrow[r] = lrow[r] * alpha + rs;
      acc0.f[r] *= alpha;
      acc1.f[r] *= alpha;
      mrow[r] = mnew;
      pbuf[flatA(row, ncol)]      = f2bf(p0);
      pbuf[flatA(row, 16 + ncol)] = f2bf(p1);
    }
    __syncthreads();                          // single-wave block: cheap DS fence
    v16bf pa = load_frag(pbuf);               // 2 x ds_load_b128
    __syncthreads();
    const size_t vtile = (size_t)(((nh << 5) + (s0 >> 5)) << 1);
    v16bf vb0 = load_frag(vP + (vtile << 9));        // d 0..15
    v16bf vb1 = load_frag(vP + ((vtile + 1) << 9));  // d 16..31
    acc0.v = wmma_bf16(pa, vb0, acc0.v);
    acc1.v = wmma_bf16(pa, vb1, acc1.v);
  }

  // Write O in A-fragment order for the mix GEMM: tiles (n, tblk) x (c-blk = head)
#pragma unroll
  for (int r = 0; r < 8; ++r) {
    const int row = (half << 3) + r;
    const float inv = 1.0f / lrow[r];
    const size_t tbase = ((size_t)((((n << 6) + qb) << 3) + h)) << 9;
    attP[tbase + flatA(row, ncol)]      = f2bf(acc0.f[r] * inv);
    attP[tbase + flatA(row, 16 + ncol)] = f2bf(acc1.f[r] * inv);
  }
}

// ------------------- Kernel 3: mix projection + bias + residual -------------------
// grid (64 tblk, 16 jblk, 8 n), 1 wave.
__global__ void __launch_bounds__(32)
mix_gemm(const unsigned short* __restrict__ attP, const unsigned short* __restrict__ wmixP,
         const float* __restrict__ b_mix, const float* __restrict__ image,
         float* __restrict__ out) {
  const int tb = blockIdx.x, jb = blockIdx.y, n = blockIdx.z;
  const int t0 = tb << 4, j0 = jb << 4;
  F8 acc; acc.v = (v8f){0.f, 0.f, 0.f, 0.f, 0.f, 0.f, 0.f, 0.f};
#pragma unroll
  for (int kc = 0; kc < NCH; kc += 32) {
    const int cb = kc >> 5;
    const int tileB = cb * 16 + jb;
    if (kc + 32 < NCH)
      __builtin_prefetch(wmixP + ((size_t)(tileB + 16) << 9), 0, 0);
    v16bf a = load_frag(attP + ((size_t)((((n << 6) + tb) << 3) + cb)) * 512);
    v16bf b = load_frag(wmixP + ((size_t)tileB << 9));
    acc.v = wmma_bf16(a, b, acc.v);
  }
  const int lane = threadIdx.x & 31, ncol = lane & 15, half = lane >> 4;
  const int j = j0 + ncol;
  const float bias = b_mix[j];
#pragma unroll
  for (int r = 0; r < 8; ++r) {
    const int t = t0 + (half << 3) + r;
    const size_t o = ((size_t)(n * NCH + j)) * SEQT + t;
    out[o] = acc.f[r] + bias + image[o];
  }
}

// ------------------- Host launch -------------------
extern "C" void kernel_launch(void* const* d_in, const int* in_sizes, int n_in,
                              void* d_out, int out_size, void* d_ws, size_t ws_size,
                              hipStream_t stream) {
  const float* image = (const float*)d_in[0];   // [8,256,32,32]
  const float* w_kqv = (const float*)d_in[1];   // [256,768]
  const float* b_kqv = (const float*)d_in[2];   // [768]
  const float* w_mix = (const float*)d_in[3];   // [256,256]
  const float* b_mix = (const float*)d_in[4];   // [256]
  float* out = (float*)d_out;

  const size_t QKV = (size_t)NB * HEADS * SEQT * HDIM;        // 2097152 elems each
  unsigned short* wkqvP = (unsigned short*)d_ws;              // 196608
  unsigned short* wmixP = wkqvP + NCH * 3 * NCH;              // 65536
  unsigned short* seqP  = wmixP + NCH * NCH;                  // 2097152
  unsigned short* qP    = seqP + QKV;
  unsigned short* kP    = qP + QKV;
  unsigned short* vP    = kP + QKV;
  unsigned short* attP  = vP + QKV;                           // 2097152

  pack_weights<<<dim3((NCH * 3 * NCH + 255) / 256), dim3(256), 0, stream>>>(
      w_kqv, w_mix, wkqvP, wmixP);

  pack_seq<<<dim3(SEQT / 16, NCH / 32, NB), dim3(32), 0, stream>>>(image, seqP);

  kqv_gemm<<<dim3(SEQT / 16, (3 * NCH) / 16, NB), dim3(32), 0, stream>>>(
      seqP, wkqvP, b_kqv, qP, kP, vP);

  attn_flash<<<dim3(SEQT / 16, HEADS, NB), dim3(32), 0, stream>>>(
      qP, kP, vP, attP);

  mix_gemm<<<dim3(SEQT / 16, NCH / 16, NB), dim3(32), 0, stream>>>(
      attP, wmixP, b_mix, image, out);
}
